// DimNetOutputBlock_24953759989851
// MI455X (gfx1250) — matrix operand order repaired and checked by reading, not compiled, not run on hardware
//
#include <hip/hip_runtime.h>

typedef _Float16 h16;
typedef __attribute__((ext_vector_type(16))) _Float16 v16h;
typedef __attribute__((ext_vector_type(8)))  _Float16 v8h;
typedef __attribute__((ext_vector_type(4)))  _Float16 v4h;
typedef __attribute__((ext_vector_type(8)))  float    v8f;

#define N_NODES     40000
#define N_EDGES     640000
#define EMB         128
#define OUT_EMB     256
#define NUM_TARGETS 12

// ---- workspace layout (bytes) ----
#define POOLED_FLOATS (N_NODES * EMB)              // 5,120,000 f32
#define WUP_OFF_B   (POOLED_FLOATS * 4)            // 20,480,000 (1KB aligned)
#define WUP_HALVES  (4 * 16 * 512)                 // K=128 -> KT=4, NT=16
#define WMLP_OFF_B  (WUP_OFF_B + WUP_HALVES * 2)
#define WMLP_HALVES (8 * 16 * 512)                 // K=256 -> KT=8, NT=16 (per layer)
#define WFIN_OFF_B  (WMLP_OFF_B + 3 * WMLP_HALVES * 2)
#define WFIN_HALVES (8 * 1 * 512)                  // K=256, N padded 12->16

// -------- zero the scatter accumulator (pooled) --------
__global__ void zero_pooled_kernel(float* __restrict__ p) {
    int i = blockIdx.x * 256 + threadIdx.x;
    p[i] = 0.0f;
}

// -------- convert fp32 weights -> f16 in WMMA B-fragment order --------
// dst index = ((kt*NT + nt)*32 + lane)*16 + j ;  element = W[k][n]
// with k = kt*32 + (lane/16)*8 + (j%8) + (j/8)*16, n = nt*16 + (lane%16)  (zero pad n>=Nsrc)
__global__ void pack_w_kernel(const float* __restrict__ src, h16* __restrict__ dst,
                              int Nsrc, int NT) {
    int idx  = blockIdx.x * 256 + threadIdx.x;
    int j    = idx & 15;
    int lane = (idx >> 4) & 31;
    int frag = idx >> 9;
    int nt   = frag % NT;
    int kt   = frag / NT;
    int k    = kt * 32 + ((lane >> 4) << 3) + (j & 7) + ((j >> 3) << 4);
    int n    = nt * 16 + (lane & 15);
    float v  = (n < Nsrc) ? src[(size_t)k * Nsrc + n] : 0.0f;
    dst[idx] = (h16)v;
}

// -------- edge stage: g = rbf@W_rbf, xe = g*x, atomic scatter-sum --------
// one wave per edge: lane owns 4 channels (float4). HBM-bound streaming of x;
// fp32 atomic adds resolve in L2 (pooled = 20.5MB << 192MB L2).
__global__ __launch_bounds__(256) void edge_scatter_kernel(
    const float* __restrict__ x, const float* __restrict__ rbf,
    const int* __restrict__ idnb, const float* __restrict__ W_rbf,
    float* __restrict__ pooled) {
    __shared__ float sW[6 * EMB];
    int tid = threadIdx.x;
    for (int i = tid; i < 6 * EMB; i += 256) sW[i] = W_rbf[i];
    __syncthreads();

    int wave = tid >> 5, lane = tid & 31;
    long e = (long)blockIdx.x * 8 + wave;

    const float* rp = rbf + e * 6;
    float r0 = rp[0], r1 = rp[1], r2 = rp[2], r3 = rp[3], r4 = rp[4], r5 = rp[5];

    int c0 = lane * 4;
    const float4 x4 = *(const float4*)(x + e * EMB + c0);

    float g[4];
#pragma unroll
    for (int q = 0; q < 4; ++q) {
        int c = c0 + q;
        g[q] = r0 * sW[c] + r1 * sW[EMB + c] + r2 * sW[2 * EMB + c] +
               r3 * sW[3 * EMB + c] + r4 * sW[4 * EMB + c] + r5 * sW[5 * EMB + c];
    }

    float* dst = pooled + (size_t)idnb[e] * EMB + c0;
    unsafeAtomicAdd(dst + 0, g[0] * x4.x);
    unsafeAtomicAdd(dst + 1, g[1] * x4.y);
    unsafeAtomicAdd(dst + 2, g[2] * x4.z);
    unsafeAtomicAdd(dst + 3, g[3] * x4.w);
}

// -------- node stage: fused pooled@W_up -> 3x silu MLP -> W_final --------
// block = 256 threads (8 waves) computes a 64-row tile. Activations live in
// LDS as f16 (64 x 264 halves, padded rows to spread banks). Each wave owns a
// 4mt x 2nt grid of D tiles so every B fragment is reused by 4 WMMAs
// (B L2 traffic /4 vs nt-major schedule). All GEMMs: v_wmma_f32_16x16x32_f16.
__global__ __launch_bounds__(256) void node_mlp_kernel(
    const float* __restrict__ pooled,
    const h16* __restrict__ wup, const h16* __restrict__ wmlp,
    const h16* __restrict__ wfin, const float* __restrict__ b_mlp,
    float* __restrict__ out) {
    __shared__ h16 sm[64][264];   // 33,792 B

    int tid = threadIdx.x;
    int m0  = blockIdx.x * 64;

    // stage 0: pooled f32 -> f16 tile in LDS (float4 in, 4-half stores out)
    for (int i = tid; i < 64 * EMB / 4; i += 256) {
        int r  = i >> 5;            // 32 float4 per row
        int c4 = (i & 31) * 4;
        float4 v = *(const float4*)(pooled + (size_t)(m0 + r) * EMB + c4);
        v4h h = { (h16)v.x, (h16)v.y, (h16)v.z, (h16)v.w };
        *(v4h*)&sm[r][c4] = h;
    }
    __syncthreads();

    int wave = tid >> 5, lane = tid & 31;
    int lo     = (lane >> 4) * 8;         // A-fragment K-run base per lane half
    int rowLo  = lane & 15;               // A row within mt tile
    int orowLo = (lane >> 4) << 3;        // C/D: M = j + 8*(lane/16)
    int ocol   = lane & 15;               // C/D: N = lane%16

    const v8f vzero = {0.f, 0.f, 0.f, 0.f, 0.f, 0.f, 0.f, 0.f};

    for (int l = 0; l < 4; ++l) {
        int KT = (l == 0) ? 4 : 8;
        const h16* wp = (l == 0) ? wup : (wmlp + (size_t)(l - 1) * WMLP_HALVES);

        v8f acc[4][2];
#pragma unroll
        for (int mt = 0; mt < 4; ++mt) {
            acc[mt][0] = vzero;
            acc[mt][1] = vzero;
        }

        for (int kt = 0; kt < KT; ++kt) {
            int kb = kt * 32;
            // two B fragments for this wave's nt pair (L2-resident weights)
            v16h b0 = *(const v16h*)(wp + ((size_t)(kt * 16 + wave * 2 + 0) * 32 + lane) * 16);
            v16h b1 = *(const v16h*)(wp + ((size_t)(kt * 16 + wave * 2 + 1) * 32 + lane) * 16);
#pragma unroll
            for (int mt = 0; mt < 4; ++mt) {
                v8h a0 = *(const v8h*)&sm[mt * 16 + rowLo][kb + lo];
                v8h a1 = *(const v8h*)&sm[mt * 16 + rowLo][kb + lo + 16];
                v16h a;
#pragma unroll
                for (int q = 0; q < 8; ++q) { a[q] = a0[q]; a[q + 8] = a1[q]; }
                acc[mt][0] = __builtin_amdgcn_wmma_f32_16x16x32_f16(
                    false, a, false, b0, (short)0, acc[mt][0], false, false);
                acc[mt][1] = __builtin_amdgcn_wmma_f32_16x16x32_f16(
                    false, a, false, b1, (short)0, acc[mt][1], false, false);
            }
        }

        __syncthreads();   // all waves done reading this layer's activations
        if (l == 0) {
#pragma unroll
            for (int mt = 0; mt < 4; ++mt)
#pragma unroll
                for (int t = 0; t < 2; ++t) {
                    int nc = (wave * 2 + t) * 16 + ocol;
#pragma unroll
                    for (int j = 0; j < 8; ++j)
                        sm[mt * 16 + orowLo + j][nc] = (h16)acc[mt][t][j];
                }
        } else {
            const float* bias = b_mlp + (size_t)(l - 1) * OUT_EMB;
#pragma unroll
            for (int t = 0; t < 2; ++t) {
                int nc = (wave * 2 + t) * 16 + ocol;
                float bv = bias[nc];
#pragma unroll
                for (int mt = 0; mt < 4; ++mt)
#pragma unroll
                    for (int j = 0; j < 8; ++j) {
                        float v = acc[mt][t][j] + bv;
                        v = v / (1.0f + __expf(-v));   // silu
                        sm[mt * 16 + orowLo + j][nc] = (h16)v;
                    }
            }
        }
        __syncthreads();   // outputs visible as next layer's inputs
    }

    // final projection: K=256, single padded N tile (12 targets), waves 0..3
    if (wave < 4) {
        int frow = wave * 16 + rowLo;
        v8f acc = vzero;
        for (int kt = 0; kt < 8; ++kt) {
            int kb = kt * 32;
            v8h a0 = *(const v8h*)&sm[frow][kb + lo];
            v8h a1 = *(const v8h*)&sm[frow][kb + lo + 16];
            v16h a;
#pragma unroll
            for (int q = 0; q < 8; ++q) { a[q] = a0[q]; a[q + 8] = a1[q]; }
            v16h b = *(const v16h*)(wfin + ((size_t)kt * 32 + lane) * 16);
            acc = __builtin_amdgcn_wmma_f32_16x16x32_f16(
                false, a, false, b, (short)0, acc, false, false);
        }
        if (ocol < NUM_TARGETS) {
            int row0 = m0 + wave * 16 + orowLo;
#pragma unroll
            for (int j = 0; j < 8; ++j)
                out[(size_t)(row0 + j) * NUM_TARGETS + ocol] = acc[j];
        }
    }
}

extern "C" void kernel_launch(void* const* d_in, const int* in_sizes, int n_in,
                              void* d_out, int out_size, void* d_ws, size_t ws_size,
                              hipStream_t stream) {
    // inputs: 0 n_atoms (unused data), 1 x, 2 rbf, 3 idnb_i, 4 W_rbf,
    //         5 W_up, 6 W_mlp, 7 b_mlp, 8 W_final
    const float* x       = (const float*)d_in[1];
    const float* rbf     = (const float*)d_in[2];
    const int*   idnb    = (const int*)d_in[3];
    const float* W_rbf   = (const float*)d_in[4];
    const float* W_up    = (const float*)d_in[5];
    const float* W_mlp   = (const float*)d_in[6];
    const float* b_mlp   = (const float*)d_in[7];
    const float* W_final = (const float*)d_in[8];
    float* out = (float*)d_out;

    char*  ws     = (char*)d_ws;
    float* pooled = (float*)ws;
    h16*   wup    = (h16*)(ws + WUP_OFF_B);
    h16*   wmlp   = (h16*)(ws + WMLP_OFF_B);
    h16*   wfin   = (h16*)(ws + WFIN_OFF_B);

    // 1) zero accumulator (every call; harness does not re-poison)
    zero_pooled_kernel<<<POOLED_FLOATS / 256, 256, 0, stream>>>(pooled);

    // 2) pack weights into f16 WMMA fragment order (tiny; L2-resident after)
    pack_w_kernel<<<WUP_HALVES / 256, 256, 0, stream>>>(W_up, wup, 256, 16);
    for (int i = 0; i < 3; ++i)
        pack_w_kernel<<<WMLP_HALVES / 256, 256, 0, stream>>>(
            W_mlp + (size_t)i * 256 * 256, wmlp + (size_t)i * WMLP_HALVES, 256, 16);
    pack_w_kernel<<<WFIN_HALVES / 256, 256, 0, stream>>>(W_final, wfin, 12, 1);

    // 3) HBM-bound edge gate + L2 fp32-atomic scatter-sum
    edge_scatter_kernel<<<N_EDGES / 8, 256, 0, stream>>>(x, rbf, idnb, W_rbf, pooled);

    // 4) fused WMMA node MLP (625 blocks of 64 rows)
    node_mlp_kernel<<<N_NODES / 64, 256, 0, stream>>>(pooled, wup, wmlp, wfin, b_mlp, out);
}
